// CausalSelfAttention_55972013802254
// MI455X (gfx1250) — matrix operand order repaired and checked by reading, not compiled, not run on hardware
//
#include <hip/hip_runtime.h>
#include <hip/hip_bf16.h>

// ---------------------------------------------------------------------------
// CausalSelfAttention forward for MI455X (gfx1250, wave32, WMMA).
// B=2, S=2048, E=1024, H=16, D=64.  All GEMMs via v_wmma_f32_16x16x32_f16.
// Flash-attention K/V tiles staged in LDS via async global->LDS copies
// (ASYNCcnt path), double-buffered.
// ---------------------------------------------------------------------------

typedef __attribute__((ext_vector_type(16))) _Float16 v16h;
typedef __attribute__((ext_vector_type(8)))  float    v8f;
typedef int v4i_vs __attribute__((vector_size(16)));   // matches builtin proto

constexpr int B_ = 2;
constexpr int S_ = 2048;
constexpr int E_ = 1024;
constexpr int H_ = 16;
constexpr int D_ = 64;
constexpr int M_ = B_ * S_;         // 4096 GEMM rows

__device__ __forceinline__ v8f wmma32(v16h a, v16h b, v8f c) {
    return __builtin_amdgcn_wmma_f32_16x16x32_f16(
        false, a, false, b, (short)0, c, false, false);
}

// ---- async global -> LDS copy (16B per lane), with portable fallback -------
#if __has_builtin(__builtin_amdgcn_global_load_async_to_lds_b128)
#define ASYNC_LDS 1
#else
#define ASYNC_LDS 0
#endif

__device__ __forceinline__ void async_copy16(const _Float16* g, _Float16* l) {
#if ASYNC_LDS
    __builtin_amdgcn_global_load_async_to_lds_b128(
        (__attribute__((address_space(1))) v4i_vs*)g,
        (__attribute__((address_space(3))) v4i_vs*)l, 0, 0);
#else
    *(int4*)l = *(const int4*)g;                    // reg-staged ds_store_b128
#endif
}

__device__ __forceinline__ void wait_async_tail() {  // allow next tile in flight
#if ASYNC_LDS
#if __has_builtin(__builtin_amdgcn_s_wait_asynccnt)
    __builtin_amdgcn_s_wait_asynccnt(4);
#else
    asm volatile("s_wait_asynccnt 4" ::: "memory");
#endif
#endif
}
__device__ __forceinline__ void wait_async_all() {
#if ASYNC_LDS
#if __has_builtin(__builtin_amdgcn_s_wait_asynccnt)
    __builtin_amdgcn_s_wait_asynccnt(0);
#else
    asm volatile("s_wait_asynccnt 0" ::: "memory");
#endif
#endif
}

// ---- fragment loaders (per documented CDNA5 VGPR layouts) ------------------
// A 16x32 f16: lanes 0-15 row M=lane, elems 0-7 = K0..7, 8-15 = K16..23;
// lanes 16-31 same rows, K8..15 / K24..31.
__device__ __forceinline__ v16h load_a_f32(const float* base, int ld, int row,
                                           int k0, int lane) {
    const float* p = base + (size_t)row * ld + k0 + ((lane >> 4) << 3);
    v16h a;
#pragma unroll
    for (int i = 0; i < 8; ++i) a[i]     = (_Float16)p[i];
#pragma unroll
    for (int i = 0; i < 8; ++i) a[i + 8] = (_Float16)p[16 + i];
    return a;
}
__device__ __forceinline__ v16h load_a_f16(const _Float16* base, int ld, int row,
                                           int k0, int lane) {
    const _Float16* p = base + (size_t)row * ld + k0 + ((lane >> 4) << 3);
    v16h a;
#pragma unroll
    for (int i = 0; i < 8; ++i) a[i]     = p[i];
#pragma unroll
    for (int i = 0; i < 8; ++i) a[i + 8] = p[16 + i];
    return a;
}
// B 32x16 f16: lane holds column N=lane&15; lanes 0-15 carry K0..7 / K16..23,
// lanes 16-31 carry K8..15 / K24..31.  `src` row index is N, column index is K.
__device__ __forceinline__ v16h load_b_f32(const float* src, int ld, int n,
                                           int k0, int lane) {
    const float* p = src + (size_t)n * ld + k0 + ((lane >> 4) << 3);
    v16h b;
#pragma unroll
    for (int i = 0; i < 8; ++i) b[i]     = (_Float16)p[i];
#pragma unroll
    for (int i = 0; i < 8; ++i) b[i + 8] = (_Float16)p[16 + i];
    return b;
}
__device__ __forceinline__ v16h load_b_f16(const _Float16* src, int ld, int n,
                                           int k0, int lane) {
    const _Float16* p = src + (size_t)n * ld + k0 + ((lane >> 4) << 3);
    v16h b;
#pragma unroll
    for (int i = 0; i < 8; ++i) b[i]     = p[i];
#pragma unroll
    for (int i = 0; i < 8; ++i) b[i + 8] = p[16 + i];
    return b;
}

// ---------------------------------------------------------------------------
// Kernel 1: fused QKV projection, out = X @ W^T + b -> f16.
// Wave tile 32(M) x 64(N): 2 A-frags x 4 B-frags -> 8 WMMAs per K-step.
//   Q,K -> [B,H,S,D] ;  V -> [B,H,D,S] (transposed for PV B-frags)
// grid = (M/32, 48/4), block = 128.
// ---------------------------------------------------------------------------
__global__ __launch_bounds__(128) void qkv_proj_kernel(
    const float* __restrict__ X,
    const float* __restrict__ qw, const float* __restrict__ qb,
    const float* __restrict__ kw, const float* __restrict__ kb,
    const float* __restrict__ vw, const float* __restrict__ vb,
    _Float16* __restrict__ Qh, _Float16* __restrict__ Kh,
    _Float16* __restrict__ Vt) {
    const int lane = threadIdx.x & 31;
    const int wv   = threadIdx.x >> 5;
    const int m0   = blockIdx.x * 32;
    const int g    = blockIdx.y * 4 + wv;        // 0..47 wave-columns of 64
    const int proj = g >> 4;                     // 16 wave-cols per projection
    const int n0   = (g & 15) * 64;

    const float* W;  const float* bias;
    if (proj == 0)      { W = qw; bias = qb; }
    else if (proj == 1) { W = kw; bias = kb; }
    else                { W = vw; bias = vb; }

    const int ar = m0 + (lane & 15);
    const int bc = n0 + (lane & 15);

    v8f c[2][4] = {};
    for (int k0 = 0; k0 < E_; k0 += 32) {
        __builtin_prefetch(X + (size_t)ar * E_ + k0 + 64, 0, 1);
        v16h a0 = load_a_f32(X, E_, ar,      k0, lane);
        v16h a1 = load_a_f32(X, E_, ar + 16, k0, lane);
#pragma unroll
        for (int j = 0; j < 4; ++j) {
            v16h bf = load_b_f32(W, E_, bc + j * 16, k0, lane);
            c[0][j] = wmma32(a0, bf, c[0][j]);
            c[1][j] = wmma32(a1, bf, c[1][j]);
        }
    }

#pragma unroll
    for (int j = 0; j < 4; ++j) {
        const int col = bc + j * 16;
        const float bv = bias[col];
        const int hh = col >> 6, dd = col & 63;
#pragma unroll
        for (int mi = 0; mi < 2; ++mi) {
#pragma unroll
            for (int r = 0; r < 8; ++r) {
                const int row = m0 + mi * 16 + ((lane >> 4) << 3) + r;
                const int bb  = row >> 11;
                const int ss  = row & (S_ - 1);
                const _Float16 hv = (_Float16)(c[mi][j][r] + bv);
                const size_t bh = (size_t)bb * H_ + hh;
                if (proj == 0)      Qh[(bh * S_ + ss) * D_ + dd] = hv;
                else if (proj == 1) Kh[(bh * S_ + ss) * D_ + dd] = hv;
                else                Vt[(bh * D_ + dd) * S_ + ss] = hv;
            }
        }
    }
}

// ---------------------------------------------------------------------------
// Kernel 2: causal flash attention, block-cooperative K/V staging.
// Block = 4 waves = 64 queries of one (b,h); key tiles of 32 (= WMMA K).
// K/V tiles double-buffered in LDS via async global->LDS; uniform trip count.
// grid = (B*H, S/64), block = 128.
// ---------------------------------------------------------------------------
__global__ __launch_bounds__(128) void flash_attn_kernel(
    const _Float16* __restrict__ Qh, const _Float16* __restrict__ Kh,
    const _Float16* __restrict__ Vt, _Float16* __restrict__ AO) {
    __shared__ alignas(16) _Float16 Kbuf[2][32 * 64];   // [key][d]
    __shared__ alignas(16) _Float16 Vbuf[2][64 * 32];   // [d][key]
    __shared__ alignas(16) _Float16 Pbuf[4][16 * 32];   // per-wave P tile

    const int tid  = threadIdx.x;
    const int lane = tid & 31;
    const int wv   = tid >> 5;
    const int bh   = blockIdx.x;
    const int b    = bh >> 4;
    const int h    = bh & 15;
    const int qb0  = blockIdx.y * 64;
    const int q0   = qb0 + wv * 16;

    const _Float16* Q  = Qh + (size_t)bh * S_ * D_;
    const _Float16* Kp = Kh + (size_t)bh * S_ * D_;
    const _Float16* Vp = Vt + (size_t)bh * D_ * S_;     // [D][S]
    _Float16* P = Pbuf[wv];

    // cooperative async stage of one 32-key K tile + V tile (4 x 16B / thread)
    auto stage_tile = [&](int ti) {
        const int buf = ti & 1;
        const int t0  = ti * 32;
        const _Float16* gk = Kp + (size_t)t0 * D_;      // contiguous 4KB
        async_copy16(gk + tid * 8,          Kbuf[buf] + tid * 8);
        async_copy16(gk + (tid + 128) * 8,  Kbuf[buf] + (tid + 128) * 8);
        const int c0 = tid, c1 = tid + 128;             // 256 16B chunks
        const int r0 = c0 >> 2, f0 = (c0 & 3) * 8;
        const int r1 = c1 >> 2, f1 = (c1 & 3) * 8;
        async_copy16(Vp + (size_t)r0 * S_ + t0 + f0, Vbuf[buf] + r0 * 32 + f0);
        async_copy16(Vp + (size_t)r1 * S_ + t0 + f1, Vbuf[buf] + r1 * 32 + f1);
    };

    // Q A-fragments (d = 0..63 in two K=32 chunks)
    const int qrow = q0 + (lane & 15);
    const v16h qa0 = load_a_f16(Q, D_, qrow, 0,  lane);
    const v16h qa1 = load_a_f16(Q, D_, qrow, 32, lane);

    float mrun[8], lrun[8];
    v8f o0 = {}, o1 = {}, o2 = {}, o3 = {};
#pragma unroll
    for (int r = 0; r < 8; ++r) { mrun[r] = -1e30f; lrun[r] = 0.0f; }

    const float scale = 0.125f;                 // 1/sqrt(64)
    const int   kend  = q0 + 16;                // this wave needs keys < kend
    const int   ntile = (qb0 + 64) / 32;        // uniform across the block

    stage_tile(0);
    for (int ti = 0; ti < ntile; ++ti) {
        if (ti + 1 < ntile) { stage_tile(ti + 1); wait_async_tail(); }
        else                { wait_async_all(); }
        __syncthreads();                        // tile ti valid for everyone

        const int t0 = ti * 32;
        if (t0 < kend) {
            const _Float16* Kt = Kbuf[ti & 1];  // [32 keys][64 d]
            const _Float16* Vc = Vbuf[ti & 1];  // [64 d][32 keys]

            // ---- S = Q K^T (16x32 = two 16x16 accumulators) ----
            v8f s0 = {}, s1 = {};
            {
                const int n = lane & 15;        // local key index
                v16h kb0 = load_b_f16(Kt, D_, n,      0,  lane);
                v16h kb1 = load_b_f16(Kt, D_, n,      32, lane);
                v16h kb2 = load_b_f16(Kt, D_, n + 16, 0,  lane);
                v16h kb3 = load_b_f16(Kt, D_, n + 16, 32, lane);
                s0 = wmma32(qa0, kb0, s0); s0 = wmma32(qa1, kb1, s0);
                s1 = wmma32(qa0, kb2, s1); s1 = wmma32(qa1, kb3, s1);
            }

            // ---- scale, causal mask, online softmax ----
            float p0v[8], p1v[8];
#pragma unroll
            for (int r = 0; r < 8; ++r) {
                const int qi = q0 + ((lane >> 4) << 3) + r;
                const int kc = t0 + (lane & 15);
                float a0 = s0[r] * scale; if (kc      > qi) a0 = -1e30f;
                float a1 = s1[r] * scale; if (kc + 16 > qi) a1 = -1e30f;
                float mx = fmaxf(a0, a1);
#pragma unroll
                for (int off = 8; off; off >>= 1)
                    mx = fmaxf(mx, __shfl_xor(mx, off, 32));
                const float mnew = fmaxf(mrun[r], mx);
                const float fac  = __expf(mrun[r] - mnew);
                const float p0   = __expf(a0 - mnew);
                const float p1   = __expf(a1 - mnew);
                float rs = p0 + p1;
#pragma unroll
                for (int off = 8; off; off >>= 1)
                    rs += __shfl_xor(rs, off, 32);
                lrun[r] = lrun[r] * fac + rs;
                mrun[r] = mnew;
                o0[r] *= fac; o1[r] *= fac; o2[r] *= fac; o3[r] *= fac;
                p0v[r] = p0; p1v[r] = p1;
            }

            // ---- re-layout P (C-tile) -> A-fragment via LDS (wave-private) --
            {
                const int rb = (lane >> 4) << 3;
                const int cc = lane & 15;
#pragma unroll
                for (int r = 0; r < 8; ++r) {
                    P[(rb + r) * 32 + cc]      = (_Float16)p0v[r];
                    P[(rb + r) * 32 + 16 + cc] = (_Float16)p1v[r];
                }
            }
            asm volatile("s_wait_dscnt 0" ::: "memory");
            const v16h pf = load_a_f16(P, 32, lane & 15, 0, lane);

            // ---- O += P @ V ----
            {
                const int dd = lane & 15;
                v16h vb0 = load_b_f16(Vc, 32, dd,      0, lane);
                v16h vb1 = load_b_f16(Vc, 32, dd + 16, 0, lane);
                v16h vb2 = load_b_f16(Vc, 32, dd + 32, 0, lane);
                v16h vb3 = load_b_f16(Vc, 32, dd + 48, 0, lane);
                o0 = wmma32(pf, vb0, o0);
                o1 = wmma32(pf, vb1, o1);
                o2 = wmma32(pf, vb2, o2);
                o3 = wmma32(pf, vb3, o3);
            }
        }
        __syncthreads();                        // done reading tile ti
    }

    // ---- normalize, store attn output as f16 [B,S,E] ----
    const int cc = lane & 15;
#pragma unroll
    for (int r = 0; r < 8; ++r) {
        const float inv = 1.0f / lrun[r];
        const int ss = q0 + ((lane >> 4) << 3) + r;
        const size_t base = ((size_t)b * S_ + ss) * E_ + h * 64 + cc;
        AO[base]      = (_Float16)(o0[r] * inv);
        AO[base + 16] = (_Float16)(o1[r] * inv);
        AO[base + 32] = (_Float16)(o2[r] * inv);
        AO[base + 48] = (_Float16)(o3[r] * inv);
    }
}

// ---------------------------------------------------------------------------
// Kernel 3: output projection, Y = AO @ o_w^T + o_b (f32), 32x64 wave tiles.
// grid = (M/32, 16/4), block = 128.
// ---------------------------------------------------------------------------
__global__ __launch_bounds__(128) void out_proj_kernel(
    const _Float16* __restrict__ AO, const float* __restrict__ ow,
    const float* __restrict__ ob, float* __restrict__ Y) {
    const int lane = threadIdx.x & 31;
    const int wv   = threadIdx.x >> 5;
    const int m0   = blockIdx.x * 32;
    const int n0   = (blockIdx.y * 4 + wv) * 64;

    const int ar = m0 + (lane & 15);
    const int bc = n0 + (lane & 15);

    v8f c[2][4] = {};
    for (int k0 = 0; k0 < E_; k0 += 32) {
        v16h a0 = load_a_f16(AO, E_, ar,      k0, lane);
        v16h a1 = load_a_f16(AO, E_, ar + 16, k0, lane);
#pragma unroll
        for (int j = 0; j < 4; ++j) {
            v16h bf = load_b_f32(ow, E_, bc + j * 16, k0, lane);
            c[0][j] = wmma32(a0, bf, c[0][j]);
            c[1][j] = wmma32(a1, bf, c[1][j]);
        }
    }

#pragma unroll
    for (int j = 0; j < 4; ++j) {
        const int col = bc + j * 16;
        const float bv = ob[col];
#pragma unroll
        for (int mi = 0; mi < 2; ++mi) {
#pragma unroll
            for (int r = 0; r < 8; ++r) {
                const int row = m0 + mi * 16 + ((lane >> 4) << 3) + r;
                Y[(size_t)row * E_ + col] = c[mi][j][r] + bv;
            }
        }
    }
}

// ---------------------------------------------------------------------------
extern "C" void kernel_launch(void* const* d_in, const int* in_sizes, int n_in,
                              void* d_out, int out_size, void* d_ws,
                              size_t ws_size, hipStream_t stream) {
    const float* X  = (const float*)d_in[0];
    const float* qw = (const float*)d_in[1];
    const float* qb = (const float*)d_in[2];
    const float* kw = (const float*)d_in[3];
    const float* kb = (const float*)d_in[4];
    const float* vw = (const float*)d_in[5];
    const float* vb = (const float*)d_in[6];
    const float* ow = (const float*)d_in[7];
    const float* ob = (const float*)d_in[8];
    float* Y = (float*)d_out;

    const size_t BSE = (size_t)B_ * S_ * E_;
    _Float16* Qh = (_Float16*)d_ws;
    _Float16* Kh = Qh + BSE;
    _Float16* Vt = Kh + BSE;
    _Float16* AO = Vt + BSE;                           // 32 MB total

    dim3 g1(M_ / 32, 12);                              // 3*1024/64 = 48 cols
    qkv_proj_kernel<<<g1, 128, 0, stream>>>(X, qw, qb, kw, kb, vw, vb,
                                            Qh, Kh, Vt);

    dim3 g2(B_ * H_, S_ / 64);                         // (32, 32)
    flash_attn_kernel<<<g2, 128, 0, stream>>>(Qh, Kh, Vt, AO);

    dim3 g3(M_ / 32, 4);                               // 1024/64 = 16 cols
    out_proj_kernel<<<g3, 128, 0, stream>>>(AO, ow, ob, Y);
}
